// PIE_8040178778148
// MI455X (gfx1250) — matrix operand order repaired and checked by reading, not compiled, not run on hardware
//
#include <hip/hip_runtime.h>
#include <hip/hip_bf16.h>

typedef __attribute__((ext_vector_type(16))) _Float16 v16h;
typedef __attribute__((ext_vector_type(8)))  float    v8f;

// ---- constants from the reference ----
static constexpr int NOUT   = 128;
static constexpr int K_EDGE = 268;   // 16*16 RBF + 12 orientation
static constexpr int K_NODE = 96;    // 6*16 RBF

// Swizzled f16 weight buffer layout in d_ws (halves):
//   edge: 9 kt * 8 nt fragments of 512 halves  -> [0, 36864)
//   node: 3 kt * 8 nt fragments of 512 halves  -> [36864, 49152)
static constexpr int EDGE_FRAG_HALVES = 9 * 8 * 512;   // 36864
static constexpr int NODE_FRAG_HALVES = 3 * 8 * 512;   // 12288
static constexpr int TOT_FRAG_HALVES  = EDGE_FRAG_HALVES + NODE_FRAG_HALVES;

// EDGE_PAIRS / NODE_PAIRS (a = src atom, b = dst atom)
static constexpr int EP_A[16] = {1,1,2,1,0,1,3,2,2,0,2,3,0,0,3,3};
static constexpr int EP_B[16] = {1,2,1,0,1,3,1,2,0,2,3,2,0,3,0,3};
static constexpr int NP_A[6]  = {1,1,1,0,0,3};
static constexpr int NP_B[6]  = {0,2,3,2,3,2};

__device__ __forceinline__ float rbf_val(float Dv, float r) {
    const float mu = r * (20.0f / 15.0f);   // linspace(0,20,16)
    const float t  = (Dv - mu) * 0.8f;      // sigma = 1.25
    return __expf(-t * t);
}

__device__ __forceinline__ void norm3(float v[3]) {
    float n = sqrtf(v[0]*v[0] + v[1]*v[1] + v[2]*v[2]);
    n = (n == 0.0f) ? 1.0f : n;
    const float inv = 1.0f / n;
    v[0] *= inv; v[1] *= inv; v[2] *= inv;
}

// ---------- one-shot: convert f32 weights -> f16 B-fragment swizzle ----------
// Fragment (kt,nt), element (l,j):  value = W[n*K + k],
//   k = kt*32 + (j<8 ? 0 : 16) + (l>=16 ? 8 : 0) + (j&7),  n = nt*16 + (l&15)   (zero-pad k >= K)
__global__ __launch_bounds__(256)
void PIE_convert_w(const float* __restrict__ edgeW,
                   const float* __restrict__ nodeW,
                   _Float16* __restrict__ ws)
{
    const int tid = blockIdx.x * blockDim.x + threadIdx.x;
    if (tid >= TOT_FRAG_HALVES) return;
    const float* W; int K; int rel;
    if (tid < EDGE_FRAG_HALVES) { W = edgeW; K = K_EDGE; rel = tid; }
    else                        { W = nodeW; K = K_NODE; rel = tid - EDGE_FRAG_HALVES; }
    const int frag   = rel >> 9;
    const int within = rel & 511;
    const int l = within >> 4, j = within & 15;
    const int kt = frag >> 3,  nt = frag & 7;
    const int k = kt * 32 + ((j < 8) ? 0 : 16) + ((l >= 16) ? 8 : 0) + (j & 7);
    const int n = nt * 16 + (l & 15);
    const float v = (k < K) ? W[n * K + k] : 0.0f;
    ws[tid] = (_Float16)v;
}

// Fused bias + LayerNorm(ddof=1) epilogue on the WMMA C/D layout.
// Output is streamed once -> non-temporal stores (keep L2 for X/idx/weights).
__device__ __forceinline__ void epilogue_ln_store(
    v8f acc[8], const float* __restrict__ bvec,
    const float* __restrict__ gain, const float* __restrict__ bias,
    float* __restrict__ out, long long rowbase, int rows_valid, int lane)
{
    const int col0 = lane & 15;
    const int half = lane >> 4;
    #pragma unroll
    for (int r = 0; r < 8; ++r) {
        const int mrow = r + half * 8;
        float x[8];
        float s = 0.0f, sq = 0.0f;
        #pragma unroll
        for (int nt = 0; nt < 8; ++nt) {
            const float v = acc[nt][r] + bvec[nt * 16 + col0];
            x[nt] = v; s += v; sq += v * v;
        }
        #pragma unroll
        for (int m = 1; m <= 8; m <<= 1) {
            s  += __shfl_xor(s,  m, 32);
            sq += __shfl_xor(sq, m, 32);
        }
        const float mu  = s * (1.0f / 128.0f);
        const float var = (sq - 128.0f * mu * mu) * (1.0f / 127.0f);
        const float sig = sqrtf(var + 1e-6f) + 1e-6f;
        const float inv = 1.0f / sig;
        if (mrow < rows_valid) {
            const long long o = rowbase + (long long)mrow * NOUT;
            #pragma unroll
            for (int nt = 0; nt < 8; ++nt) {
                const int col = nt * 16 + col0;
                __builtin_nontemporal_store(
                    gain[col] * (x[nt] - mu) * inv + bias[col], out + o + col);
            }
        }
    }
}

// ------------- per-tile edge work: features -> WMMA -> LN -> store -----------
__device__ __forceinline__ void edge_tile(
    const _Float16* sW /* LDS: kt 0..7, 64 frags */,
    const _Float16* __restrict__ gTail /* global: kt=8, 8 frags */,
    const float* __restrict__ X, const int* __restrict__ idx,
    const float* __restrict__ bvec, const float* __restrict__ gain,
    const float* __restrict__ bias, float* __restrict__ out,
    int base, int E, int N, int lane)
{
    const int mrow = lane & 15;
    int e = base + mrow;
    if (e >= E) e = E - 1;

    // edge indices are streamed exactly once -> non-temporal loads
    const int src = __builtin_nontemporal_load(idx + e);
    const int dst = __builtin_nontemporal_load(idx + E + e);

    const float* ps = X + (size_t)src * 12;
    const float* pd = X + (size_t)dst * 12;
    float xs[4][3], xd[4][3];
    #pragma unroll
    for (int i = 0; i < 4; ++i)
        #pragma unroll
        for (int c = 0; c < 3; ++c) { xs[i][c] = ps[i*3+c]; xd[i][c] = pd[i*3+c]; }

    float D[16];
    #pragma unroll
    for (int p = 0; p < 16; ++p) {
        float dd = 1e-6f;
        #pragma unroll
        for (int c = 0; c < 3; ++c) {
            const float d = xs[EP_A[p]][c] - xd[EP_B[p]][c];
            dd += d * d;
        }
        D[p] = sqrtf(dd);
    }

    // orientation features (12): local frame of src residue
    float u0[3], u1[3];
    #pragma unroll
    for (int c = 0; c < 3; ++c) { u0[c] = xs[1][c] - xs[0][c]; u1[c] = xs[2][c] - xs[1][c]; }
    norm3(u0); norm3(u1);
    float n0[3] = { u0[1]*u1[2] - u0[2]*u1[1],
                    u0[2]*u1[0] - u0[0]*u1[2],
                    u0[0]*u1[1] - u0[1]*u1[0] };
    norm3(n0);
    float b1[3];
    #pragma unroll
    for (int c = 0; c < 3; ++c) b1[c] = u0[c] - u1[c];
    norm3(b1);
    const float cr[3] = { b1[1]*n0[2] - b1[2]*n0[1],
                          b1[2]*n0[0] - b1[0]*n0[2],
                          b1[0]*n0[1] - b1[1]*n0[0] };
    const float qz = (src == N - 1) ? 0.0f : 1.0f;

    float dir[12];
    {
        constexpr int ord[4] = {1, 0, 2, 3};
        #pragma unroll
        for (int a = 0; a < 4; ++a) {
            float dX[3], du[3];
            #pragma unroll
            for (int c = 0; c < 3; ++c) dX[c] = xd[ord[a]][c] - xs[0][c];
            #pragma unroll
            for (int i = 0; i < 3; ++i)
                du[i] = qz * (b1[i]*dX[0] + n0[i]*dX[1] + cr[i]*dX[2]);
            float nn = sqrtf(du[0]*du[0] + du[1]*du[1] + du[2]*du[2]);
            nn = (nn == 0.0f) ? 1.0f : nn;
            const float inv = 1.0f / nn;
            #pragma unroll
            for (int i = 0; i < 3; ++i) dir[a*3 + i] = du[i] * inv;
        }
    }

    v8f acc[8];
    {
        const v8f z = {};
        #pragma unroll
        for (int nt = 0; nt < 8; ++nt) acc[nt] = z;
    }
    const float rbase  = (lane >= 16) ? 8.0f : 0.0f;
    const bool  hiHalf = (lane >= 16);

    #pragma unroll
    for (int kt = 0; kt < 9; ++kt) {
        v16h a;
        if (kt < 8) {
            const int p0 = 2 * kt, p1 = 2 * kt + 1;
            #pragma unroll
            for (int j = 0; j < 8; ++j) {
                const float r = rbase + (float)j;
                a[j]     = (_Float16)rbf_val(D[p0], r);
                a[j + 8] = (_Float16)rbf_val(D[p1], r);
            }
        } else {
            #pragma unroll
            for (int j = 0; j < 8; ++j) {
                const float lo = dir[j];
                const float hi = (8 + j < 12) ? dir[8 + j] : 0.0f;
                a[j]     = (_Float16)(hiHalf ? hi : lo);
                a[j + 8] = (_Float16)0.0f;
            }
        }
        #pragma unroll
        for (int nt = 0; nt < 8; ++nt) {
            v16h b;
            if (kt < 8) b = *(const v16h*)(sW    + ((kt * 8 + nt) << 9) + lane * 16);
            else        b = *(const v16h*)(gTail + (nt << 9)            + lane * 16);
            acc[nt] = __builtin_amdgcn_wmma_f32_16x16x32_f16(
                false, a, false, b, (short)0, acc[nt], false, false);
        }
    }

    int rv = E - base; rv = rv > 16 ? 16 : rv;
    epilogue_ln_store(acc, bvec, gain, bias, out, (long long)base * NOUT, rv, lane);
}

// -------- edge kernel: 4 waves/block, 8 tiles/block (512 edges/block) --------
static constexpr int TILES_PER_BLOCK = 8;

__global__ __launch_bounds__(128)
void PIE_edge_kernel(const float* __restrict__ X, const int* __restrict__ idx,
                     const _Float16* __restrict__ wsEdge,
                     const float* __restrict__ bvec,
                     const float* __restrict__ gain, const float* __restrict__ bias,
                     float* __restrict__ out, int E, int N)
{
    __shared__ _Float16 sW[64 * 512];   // kt 0..7 fragments, 65536 bytes

    // cooperative stage of the hot B-fragments: 2048 x 32B chunks
    #pragma unroll 4
    for (int i = threadIdx.x; i < 2048; i += 128)
        *(v16h*)(sW + i * 16) = *(const v16h*)(wsEdge + i * 16);
    __syncthreads();

    const _Float16* gTail = wsEdge + 64 * 512;   // kt=8 fragments (L2-resident)
    const int lane = threadIdx.x & 31;
    const int wave = threadIdx.x >> 5;

    #pragma unroll 1
    for (int t = 0; t < TILES_PER_BLOCK; ++t) {
        const int base = (blockIdx.x * TILES_PER_BLOCK + t) * 64 + wave * 16;
        if (base < E)
            edge_tile(sW, gTail, X, idx, bvec, gain, bias, out, base, E, N, lane);
    }
}

// ---------------- node kernel: 4 waves/block, 16 nodes/wave -----------------
__global__ __launch_bounds__(128)
void PIE_node_kernel(const float* __restrict__ X,
                     const _Float16* __restrict__ wsNode,
                     const float* __restrict__ bvec,
                     const float* __restrict__ gain, const float* __restrict__ bias,
                     float* __restrict__ out, int N)
{
    const int lane = threadIdx.x & 31;
    const int wave = threadIdx.x >> 5;
    const int base = blockIdx.x * 64 + wave * 16;
    const int mrow = lane & 15;
    int node = base + mrow;
    if (node >= N) node = N - 1;

    const float* ps = X + (size_t)node * 12;
    float xs[4][3];
    #pragma unroll
    for (int i = 0; i < 4; ++i)
        #pragma unroll
        for (int c = 0; c < 3; ++c) xs[i][c] = ps[i*3+c];

    float D[6];
    #pragma unroll
    for (int p = 0; p < 6; ++p) {
        float dd = 1e-6f;
        #pragma unroll
        for (int c = 0; c < 3; ++c) {
            const float d = xs[NP_A[p]][c] - xs[NP_B[p]][c];
            dd += d * d;
        }
        D[p] = sqrtf(dd);
    }

    v8f acc[8];
    {
        const v8f z = {};
        #pragma unroll
        for (int nt = 0; nt < 8; ++nt) acc[nt] = z;
    }
    const float rbase = (lane >= 16) ? 8.0f : 0.0f;

    #pragma unroll
    for (int kt = 0; kt < 3; ++kt) {
        const int p0 = 2 * kt, p1 = 2 * kt + 1;
        v16h a;
        #pragma unroll
        for (int j = 0; j < 8; ++j) {
            const float r = rbase + (float)j;
            a[j]     = (_Float16)rbf_val(D[p0], r);
            a[j + 8] = (_Float16)rbf_val(D[p1], r);
        }
        #pragma unroll
        for (int nt = 0; nt < 8; ++nt) {
            const v16h b = *(const v16h*)(wsNode + ((kt * 8 + nt) << 9) + lane * 16);
            acc[nt] = __builtin_amdgcn_wmma_f32_16x16x32_f16(
                false, a, false, b, (short)0, acc[nt], false, false);
        }
    }

    int rv = N - base; rv = rv > 16 ? 16 : rv;
    epilogue_ln_store(acc, bvec, gain, bias, out, (long long)base * NOUT, rv, lane);
}

extern "C" void kernel_launch(void* const* d_in, const int* in_sizes, int n_in,
                              void* d_out, int out_size, void* d_ws, size_t ws_size,
                              hipStream_t stream) {
    const float* X      = (const float*)d_in[0];
    const int*   Ein    = (const int*)  d_in[1];
    const int*   Eex    = (const int*)  d_in[2];
    const float* nodeW  = (const float*)d_in[3];
    const float* nodeB  = (const float*)d_in[4];
    const float* edgeW  = (const float*)d_in[5];
    const float* edgeB  = (const float*)d_in[6];
    const float* gain_n = (const float*)d_in[7];
    const float* bias_n = (const float*)d_in[8];
    const float* gain_e = (const float*)d_in[9];
    const float* bias_e = (const float*)d_in[10];

    const int N = in_sizes[0] / 12;   // (N,4,3)
    const int E = in_sizes[1] / 2;    // (2,E)

    float* hV   = (float*)d_out;
    float* hEin = hV   + (size_t)N * NOUT;
    float* hEex = hEin + (size_t)E * NOUT;

    _Float16* wsHalf = (_Float16*)d_ws;
    _Float16* wsEdge = wsHalf;                       // 36864 halves
    _Float16* wsNode = wsHalf + EDGE_FRAG_HALVES;    // 12288 halves

    // one-shot weight convert + swizzle (stream-ordered before the GEMMs)
    PIE_convert_w<<<dim3((TOT_FRAG_HALVES + 255) / 256), dim3(256), 0, stream>>>(
        edgeW, nodeW, wsHalf);

    const dim3 blk(128);
    PIE_node_kernel<<<dim3((N + 63) / 64), blk, 0, stream>>>(
        X, wsNode, nodeB, gain_n, bias_n, hV, N);

    const int edgeBlocks = (E + 64 * TILES_PER_BLOCK - 1) / (64 * TILES_PER_BLOCK);
    PIE_edge_kernel<<<dim3(edgeBlocks), blk, 0, stream>>>(
        X, Ein, wsEdge, edgeB, gain_e, bias_e, hEin, E, N);
    PIE_edge_kernel<<<dim3(edgeBlocks), blk, 0, stream>>>(
        X, Eex, wsEdge, edgeB, gain_e, bias_e, hEex, E, N);
}
